// Attention_59914793779425
// MI455X (gfx1250) — matrix operand order repaired and checked by reading, not compiled, not run on hardware
//
#include <hip/hip_runtime.h>
#include <hip/hip_bf16.h>
#include <stdint.h>

// ---------------------------------------------------------------------------
// MHA forward for MI455X (gfx1250, wave32).  All matmuls go through
// V_WMMA_F32_16X16X32_BF16 (f32 accumulate).  ~69 GFLOP vs tens of MB of
// traffic -> compute bound -> WMMA path, bf16 operands.  Tile staging uses
// the CDNA5 data-movement paths: TDM (tensor_load_to_lds, TENSORcnt,
// double-buffered) for attention K tiles, GLOBAL_LOAD_ASYNC_TO_LDS_B128
// (ASYNCcnt, double-buffered) for projection A tiles.
// ---------------------------------------------------------------------------

typedef __bf16 bf16_t;
typedef __attribute__((ext_vector_type(16))) __bf16 bf16x16;
typedef __attribute__((ext_vector_type(8)))  __bf16 bf16x8;
typedef __attribute__((ext_vector_type(8)))  float  f32x8;
typedef __attribute__((ext_vector_type(4))) unsigned int u32x4;
typedef __attribute__((ext_vector_type(8))) int i32x8;
typedef __attribute__((ext_vector_type(4))) int i32x4;

__device__ __forceinline__ f32x8 wmma_bf16(bf16x16 a, bf16x16 b, f32x8 c) {
  // (neg_a, A, neg_b, B, c_mod, C, reuse_a, reuse_b)
  return __builtin_amdgcn_wmma_f32_16x16x32_bf16(false, a, false, b, (short)0, c,
                                                 false, false);
}

// Low 32 bits of a generic pointer to LDS == LDS byte address (ISA 10.2).
__device__ __forceinline__ uint32_t lds_addr_of(const void* p) {
  return (uint32_t)(uintptr_t)p;
}

// GLOBAL_LOAD_ASYNC_TO_LDS_B128: DMA 16 bytes global -> LDS, tracked by
// ASYNCcnt.  VDST = LDS byte address VGPR, VADDR = 64-bit global address.
__device__ __forceinline__ void async_copy_b128(uint32_t lds_byte_addr,
                                                const void* gptr) {
  asm volatile("global_load_async_to_lds_b128 %0, %1, off"
               :
               : "v"(lds_byte_addr), "v"(gptr)
               : "memory");
}
__device__ __forceinline__ void wait_async_le2() {
  asm volatile("s_wait_asynccnt 2" ::: "memory");  // oldest 2 (current tile) done
}

// ---------------------------------------------------------------------------
// Tensor Data Mover: 2D tile load global -> LDS with optional LDS padding.
// Builtin arity differs per toolchain (probe doc): 5 args on ROCm7.2/clang-22,
// 6 args on amdgpu-toolchain/clang-23; prefer a d2 (2-group) form if present.
// ---------------------------------------------------------------------------
__device__ __forceinline__ void tdm_load_tile_2d(
    uint32_t lds_addr, const void* gaddr,
    uint32_t tensor_d0, uint32_t tensor_d1,   // in elements (data_size units)
    uint32_t tile_d0, uint32_t tile_d1,
    uint32_t stride0,                         // elements between dim1 lines
    uint32_t pad_enable, uint32_t pad_interval_code, uint32_t pad_amount_code) {
  const uint64_t ga = (uint64_t)(uintptr_t)gaddr;
  u32x4 g0;
  g0.x = 1u;                                   // count=1, user descriptor
  g0.y = lds_addr;                             // D#.lds_addr
  g0.z = (uint32_t)ga;                         // global_addr[31:0]
  g0.w = (uint32_t)((ga >> 32) & 0x1FFFFFFu)   // global_addr[56:32]
         | (2u << 30);                         // type = 2 ("image")
  i32x8 g1;
  g1[0] = (int)((1u << 16)                     // data_size = 1 -> 2 bytes
                | (pad_enable << 20)
                | (pad_interval_code << 22)
                | (pad_amount_code << 25));
  g1[1] = (int)((tensor_d0 & 0xFFFFu) << 16);                      // dim0 lo
  g1[2] = (int)(((tensor_d0 >> 16) & 0xFFFFu) |
                ((tensor_d1 & 0xFFFFu) << 16));                    // dim0 hi | dim1 lo
  g1[3] = (int)(((tensor_d1 >> 16) & 0xFFFFu) |
                ((tile_d0 & 0xFFFFu) << 16));                      // dim1 hi | tile0
  g1[4] = (int)(tile_d1 & 0xFFFFu);                                // tile1 | tile2=0
  g1[5] = (int)stride0;                                            // dim0_stride lo
  g1[6] = 0;
  g1[7] = 0;
  const i32x4 z4 = {0, 0, 0, 0};
  const i32x8 z8 = {0, 0, 0, 0, 0, 0, 0, 0};
#if __has_builtin(__builtin_amdgcn_tensor_load_to_lds_d2)
  (void)z4; (void)z8;
  __builtin_amdgcn_tensor_load_to_lds_d2((i32x4)g0, g1, 0);
#elif defined(__clang_major__) && (__clang_major__ < 23)
  (void)z8;
  __builtin_amdgcn_tensor_load_to_lds(g0, g1, z4, z4, 0);
#else
  __builtin_amdgcn_tensor_load_to_lds(g0, g1, z4, z4, z8, 0);
#endif
}

// A-matrix fragment (16x32 MxK).  rowptr -> element [m][k0] of a row-major
// array (K contiguous).  ISA layout: lane<16 holds K {0..7,16..23} of row m,
// lane>=16 holds K {8..15,24..31}.
__device__ __forceinline__ bf16x16 load_a_frag(const bf16_t* rowptr, int half) {
  bf16x8 lo = *(const bf16x8*)(rowptr + half * 8);
  bf16x8 hi = *(const bf16x8*)(rowptr + 16 + half * 8);
  bf16x16 f;
#pragma unroll
  for (int i = 0; i < 8; ++i) { f[i] = lo[i]; f[i + 8] = hi[i]; }
  return f;
}

// B-matrix fragment (32x16 KxN).  colptr -> element [n][k0] of an [N][K]
// row-major array (K contiguous).  Same layout as A: lane<16 holds
// K {0..7,16..23} of column n, lane>=16 holds K {8..15,24..31}.
__device__ __forceinline__ bf16x16 load_b_frag(const bf16_t* colptr, int half) {
  bf16x8 lo = *(const bf16x8*)(colptr + half * 8);
  bf16x8 hi = *(const bf16x8*)(colptr + 16 + half * 8);
  bf16x16 f;
#pragma unroll
  for (int i = 0; i < 8; ++i) { f[i] = lo[i]; f[i + 8] = hi[i]; }
  return f;
}

// ---------------------------------------------------------------------------
// Kernel 1: qkv = x @ Wqkv^T   (M=4096, N=3072, K=1024), fp32 in, bf16 out
// scattered into [3][B][H][N][D]; q pre-scaled by 1/sqrt(D).
// ---------------------------------------------------------------------------
__global__ __launch_bounds__(256) void qkv_gemm(const float* __restrict__ x,
                                                const float* __restrict__ w,
                                                bf16_t* __restrict__ qkv) {
  __shared__ bf16_t at[128 * 40];
  __shared__ bf16_t btile[128 * 40];
  const int tid = threadIdx.x, wave = tid >> 5, lane = tid & 31;
  const int half = lane >> 4, lm = lane & 15;
  const int bm = blockIdx.y * 128, bn = blockIdx.x * 128;
  const int wm = (wave >> 1) * 32, wn = (wave & 1) * 64;
  const int lr = tid >> 1, lc = (tid & 1) * 16;

  f32x8 acc[2][4];
#pragma unroll
  for (int i = 0; i < 2; ++i)
#pragma unroll
    for (int j = 0; j < 4; ++j)
#pragma unroll
      for (int r = 0; r < 8; ++r) acc[i][j][r] = 0.f;

  for (int k0 = 0; k0 < 1024; k0 += 32) {
    const float4* xs = (const float4*)(x + (size_t)(bm + lr) * 1024 + k0 + lc);
    const float4* ws = (const float4*)(w + (size_t)(bn + lr) * 1024 + k0 + lc);
    bf16_t* ad = at + lr * 40 + lc;
    bf16_t* bd = btile + lr * 40 + lc;
    if (k0 + 32 < 1024) {
      __builtin_prefetch(x + (size_t)(bm + lr) * 1024 + k0 + 32 + lc, 0, 0);
      __builtin_prefetch(w + (size_t)(bn + lr) * 1024 + k0 + 32 + lc, 0, 0);
    }
#pragma unroll
    for (int i = 0; i < 4; ++i) {
      float4 fa = xs[i]; float4 fb = ws[i];
      ad[i * 4 + 0] = (bf16_t)fa.x; ad[i * 4 + 1] = (bf16_t)fa.y;
      ad[i * 4 + 2] = (bf16_t)fa.z; ad[i * 4 + 3] = (bf16_t)fa.w;
      bd[i * 4 + 0] = (bf16_t)fb.x; bd[i * 4 + 1] = (bf16_t)fb.y;
      bd[i * 4 + 2] = (bf16_t)fb.z; bd[i * 4 + 3] = (bf16_t)fb.w;
    }
    __syncthreads();
    bf16x16 af[2], bfr[4];
#pragma unroll
    for (int i = 0; i < 2; ++i) af[i] = load_a_frag(at + (wm + i * 16 + lm) * 40, half);
#pragma unroll
    for (int j = 0; j < 4; ++j) bfr[j] = load_b_frag(btile + (wn + j * 16 + lm) * 40, half);
#pragma unroll
    for (int i = 0; i < 2; ++i)
#pragma unroll
      for (int j = 0; j < 4; ++j) acc[i][j] = wmma_bf16(af[i], bfr[j], acc[i][j]);
    __syncthreads();
  }

  // C/D layout: VGPR r -> row (r + 8*half), column = lane&15 of the tile.
#pragma unroll
  for (int i = 0; i < 2; ++i) {
#pragma unroll
    for (int j = 0; j < 4; ++j) {
      const int col = bn + wn + j * 16 + lm;
      const int which = col >> 10, c = col & 1023, hh = c >> 6, d = c & 63;
      const float s = (which == 0) ? 0.125f : 1.0f;  // fold softmax scale into q
#pragma unroll
      for (int r = 0; r < 8; ++r) {
        const int row = bm + wm + i * 16 + half * 8 + r;
        const int b = row >> 11, n = row & 2047;
        const size_t idx =
            ((size_t)which * 32 + (size_t)(b * 16 + hh)) * (2048 * 64) +
            (size_t)n * 64 + d;
        qkv[idx] = (bf16_t)(acc[i][j][r] * s);
      }
    }
  }
}

// ---------------------------------------------------------------------------
// Kernel 2: flash attention per (b,h).  Block: 128 query rows, 8 waves,
// each wave owns 16 rows.  Key loop in blocks of 128.  K tiles are DMA'd by
// the Tensor Data Mover into a double buffer with hardware LDS padding
// (stride 64 -> 72 elems: pad_interval code 4 = 32 DWORDs, amount code 3 = 4).
// ---------------------------------------------------------------------------
__global__ __launch_bounds__(256) void attn_fwd(const bf16_t* __restrict__ q,
                                                const bf16_t* __restrict__ k,
                                                const bf16_t* __restrict__ v,
                                                bf16_t* __restrict__ o_out) {
  __shared__ bf16_t ktile[2 * 128 * 72];    // [buf][key][d], stride 72 (TDM pad)
  __shared__ bf16_t vtile[64 * 136];        // [d][key], transposed
  __shared__ bf16_t pstage[8][16 * 40];     // per-wave P transpose stage (16x32)
  const int tid = threadIdx.x, wave = tid >> 5, lane = tid & 31;
  const int half = lane >> 4, lm = lane & 15;
  const int b = blockIdx.z, h = blockIdx.y;
  const int m0 = blockIdx.x * 128;
  const size_t bh = ((size_t)(b * 16 + h)) * (2048 * 64);
  const bf16_t* qp = q + bh;
  const bf16_t* kp = k + bh;
  const bf16_t* vp = v + bh;

  // Q fragments for this wave's 16 rows (A layout, D=64 -> two k-steps)
  const int qrow = m0 + wave * 16 + lm;
  const bf16x16 qf0 = load_a_frag(qp + (size_t)qrow * 64 + 0, half);
  const bf16x16 qf1 = load_a_frag(qp + (size_t)qrow * 64 + 32, half);

  f32x8 o[4];
  float rm[8], rl[8];
#pragma unroll
  for (int j = 0; j < 8; ++j) { rm[j] = -3.0e38f; rl[j] = 0.f; }
#pragma unroll
  for (int t = 0; t < 4; ++t)
#pragma unroll
    for (int j = 0; j < 8; ++j) o[t][j] = 0.f;

  const int lr = tid >> 1, lcb = (tid & 1) * 32;

  // Prologue: TDM kicks tile 0 into buffer 0.
  if (wave == 0) {
    tdm_load_tile_2d(lds_addr_of(ktile), kp,
                     /*tensor_d0=*/64, /*tensor_d1=*/2048,
                     /*tile_d0=*/64, /*tile_d1=*/128, /*stride0=*/64,
                     /*pad_enable=*/1, /*pad_interval=*/4, /*pad_amount=*/3);
  }
  int buf = 0;

  for (int kb = 0; kb < 2048; kb += 128) {
    if (wave == 0) {
      // Prefetch next K tile (wrapping keeps the schedule uniform; the wrap
      // tile is never consumed).  TDM completes in-order per wave, so
      // tensorcnt<=1 proves the current tile is resident.
      const int nkb = (kb + 128) & 2047;
      tdm_load_tile_2d(lds_addr_of(ktile + (buf ^ 1) * (128 * 72)),
                       kp + (size_t)nkb * 64,
                       64, 2048, 64, 128, 64, 1, 4, 3);
      __builtin_amdgcn_s_wait_tensorcnt(1);
    }
    {  // V tile transposed store: vtile[d][key]
      const bf16_t* src = vp + (size_t)(kb + lr) * 64 + lcb;
#pragma unroll
      for (int i = 0; i < 32; ++i) vtile[(lcb + i) * 136 + lr] = src[i];
    }
    __syncthreads();
    const bf16_t* kt = ktile + buf * (128 * 72);

    // S = Q K^T  (16 x 128 per wave; logits already scaled via q)
    f32x8 s[8];
#pragma unroll
    for (int t = 0; t < 8; ++t) {
      f32x8 z;
#pragma unroll
      for (int r = 0; r < 8; ++r) z[r] = 0.f;
      bf16x16 kf0 = load_b_frag(kt + (t * 16 + lm) * 72 + 0, half);
      bf16x16 kf1 = load_b_frag(kt + (t * 16 + lm) * 72 + 32, half);
      z = wmma_bf16(qf0, kf0, z);
      z = wmma_bf16(qf1, kf1, z);
      s[t] = z;
    }

    // Online softmax.  Row j lives in VGPR j across the 16 lanes of each
    // half-wave; xor masks 1..8 stay inside the half.
    float nm[8];
#pragma unroll
    for (int j = 0; j < 8; ++j) {
      float mx = s[0][j];
#pragma unroll
      for (int t = 1; t < 8; ++t) mx = fmaxf(mx, s[t][j]);
#pragma unroll
      for (int off = 1; off < 16; off <<= 1) mx = fmaxf(mx, __shfl_xor(mx, off, 32));
      nm[j] = fmaxf(rm[j], mx);
    }
#pragma unroll
    for (int j = 0; j < 8; ++j) {
      const float corr = __expf(rm[j] - nm[j]);
      rl[j] *= corr;
#pragma unroll
      for (int t = 0; t < 4; ++t) o[t][j] *= corr;
      rm[j] = nm[j];
    }
#pragma unroll
    for (int t = 0; t < 8; ++t)
#pragma unroll
      for (int j = 0; j < 8; ++j) s[t][j] = __expf(s[t][j] - rm[j]);
#pragma unroll
    for (int j = 0; j < 8; ++j) {
      float sum = 0.f;
#pragma unroll
      for (int t = 0; t < 8; ++t) sum += s[t][j];
#pragma unroll
      for (int off = 1; off < 16; off <<= 1) sum += __shfl_xor(sum, off, 32);
      rl[j] += sum;
    }

    // O += P V.  Transpose P (C layout -> A layout) 16x32 at a time through
    // per-wave LDS; same-wave DS ops are in-order, so no barrier needed.
    bf16_t* ps = pstage[wave];
#pragma unroll
    for (int kk = 0; kk < 4; ++kk) {
#pragma unroll
      for (int j = 0; j < 8; ++j) {
        ps[(half * 8 + j) * 40 + lm]      = (bf16_t)s[2 * kk][j];
        ps[(half * 8 + j) * 40 + 16 + lm] = (bf16_t)s[2 * kk + 1][j];
      }
      bf16x16 pf = load_a_frag(ps + lm * 40, half);
#pragma unroll
      for (int t = 0; t < 4; ++t) {
        bf16x16 vf = load_b_frag(vtile + (t * 16 + lm) * 136 + kk * 32, half);
        o[t] = wmma_bf16(pf, vf, o[t]);
      }
    }
    __syncthreads();
    buf ^= 1;
  }

  // Epilogue: divide by row sum, store bf16 at [b*2048+n][h*64+d].
#pragma unroll
  for (int j = 0; j < 8; ++j) {
    const float inv = 1.0f / rl[j];
    const int n = m0 + wave * 16 + half * 8 + j;
#pragma unroll
    for (int t = 0; t < 4; ++t) {
      const int d = t * 16 + lm;
      o_out[((size_t)(b * 2048 + n)) * 1024 + h * 64 + d] = (bf16_t)(o[t][j] * inv);
    }
  }
}

// ---------------------------------------------------------------------------
// Kernel 3: out = attn @ Wproj^T + bproj   (M=4096, N=1024, K=1024).
// A tiles (already bf16) are double-buffered via GLOBAL_LOAD_ASYNC_TO_LDS_B128;
// the fp32->bf16 B-tile convert overlaps the DMA.
// ---------------------------------------------------------------------------
__global__ __launch_bounds__(256) void proj_gemm(const bf16_t* __restrict__ a,
                                                 const float* __restrict__ w,
                                                 const float* __restrict__ bias,
                                                 float* __restrict__ out) {
  __shared__ bf16_t at[2 * 128 * 40];
  __shared__ bf16_t btile[128 * 40];
  const int tid = threadIdx.x, wave = tid >> 5, lane = tid & 31;
  const int half = lane >> 4, lm = lane & 15;
  const int bm = blockIdx.y * 128, bn = blockIdx.x * 128;
  const int wm = (wave >> 1) * 32, wn = (wave & 1) * 64;
  const int lr = tid >> 1, lc = (tid & 1) * 16;

  f32x8 acc[2][4];
#pragma unroll
  for (int i = 0; i < 2; ++i)
#pragma unroll
    for (int j = 0; j < 4; ++j)
#pragma unroll
      for (int r = 0; r < 8; ++r) acc[i][j][r] = 0.f;

  // Prologue: async A tile 0 into buffer 0 (2 x 16B per thread).
  {
    const uint32_t la = lds_addr_of(at + lr * 40 + lc);
    const bf16_t* g = a + (size_t)(bm + lr) * 1024 + lc;
    async_copy_b128(la, g);
    async_copy_b128(la + 16, g + 8);
  }
  int buf = 0;

  for (int k0 = 0; k0 < 1024; k0 += 32) {
    {  // Prefetch next A tile (wrapping; the wrap tile is never consumed).
      const int nk = (k0 + 32) & 1023;
      const uint32_t la = lds_addr_of(at + (buf ^ 1) * (128 * 40) + lr * 40 + lc);
      const bf16_t* g = a + (size_t)(bm + lr) * 1024 + nk + lc;
      async_copy_b128(la, g);
      async_copy_b128(la + 16, g + 8);
    }
    const float4* ws = (const float4*)(w + (size_t)(bn + lr) * 1024 + k0 + lc);
    bf16_t* bd = btile + lr * 40 + lc;
#pragma unroll
    for (int i = 0; i < 4; ++i) {
      float4 fb = ws[i];
      bd[i * 4 + 0] = (bf16_t)fb.x; bd[i * 4 + 1] = (bf16_t)fb.y;
      bd[i * 4 + 2] = (bf16_t)fb.z; bd[i * 4 + 3] = (bf16_t)fb.w;
    }
    // Async loads complete in-order: cnt<=2 means the current tile's two
    // transfers have landed; the two outstanding belong to the next tile.
    wait_async_le2();
    __syncthreads();
    const bf16_t* ac = at + buf * (128 * 40);
    bf16x16 af[2], bfr[4];
#pragma unroll
    for (int i = 0; i < 2; ++i) af[i] = load_a_frag(ac + (wm + i * 16 + lm) * 40, half);
#pragma unroll
    for (int j = 0; j < 4; ++j) bfr[j] = load_b_frag(btile + (wn + j * 16 + lm) * 40, half);
#pragma unroll
    for (int i = 0; i < 2; ++i)
#pragma unroll
      for (int j = 0; j < 4; ++j) acc[i][j] = wmma_bf16(af[i], bfr[j], acc[i][j]);
    __syncthreads();
    buf ^= 1;
  }

#pragma unroll
  for (int i = 0; i < 2; ++i) {
#pragma unroll
    for (int j = 0; j < 4; ++j) {
      const int col = bn + wn + j * 16 + lm;
      const float bv = bias[col];
#pragma unroll
      for (int r = 0; r < 8; ++r) {
        const int row = bm + wm + i * 16 + half * 8 + r;
        out[(size_t)row * 1024 + col] = acc[i][j][r] + bv;
      }
    }
  }
}

// ---------------------------------------------------------------------------
extern "C" void kernel_launch(void* const* d_in, const int* in_sizes, int n_in,
                              void* d_out, int out_size, void* d_ws, size_t ws_size,
                              hipStream_t stream) {
  (void)in_sizes; (void)n_in; (void)out_size; (void)ws_size;
  const float* x     = (const float*)d_in[0];   // [2,2048,1024]
  const float* Wqkv  = (const float*)d_in[1];   // [3072,1024]
  const float* Wproj = (const float*)d_in[2];   // [1024,1024]
  const float* bproj = (const float*)d_in[3];   // [1024]
  float* out = (float*)d_out;                   // [2,2048,1024] fp32

  // Workspace: qkv bf16 [3][2][16][2048][64] (24 MB) + attn bf16 [4096][1024] (8 MB)
  bf16_t* qkv_ws  = (bf16_t*)d_ws;
  const size_t per_tensor = (size_t)2 * 16 * 2048 * 64;  // 4,194,304
  bf16_t* attn_ws = qkv_ws + 3 * per_tensor;
  const bf16_t* q = qkv_ws;
  const bf16_t* k = qkv_ws + per_tensor;
  const bf16_t* v = qkv_ws + 2 * per_tensor;

  qkv_gemm<<<dim3(24, 32), 256, 0, stream>>>(x, Wqkv, qkv_ws);
  attn_fwd<<<dim3(16, 16, 2), 256, 0, stream>>>(q, k, v, attn_ws);
  proj_gemm<<<dim3(8, 32), 256, 0, stream>>>(attn_ws, Wproj, bproj, out);
}
